// LeakySNNMNIST_79585743995305
// MI455X (gfx1250) — compile-verified, matrix operand order checked by reading, not run on hardware
//
#include <hip/hip_runtime.h>

// CDNA5 / gfx1250 fused spiking-MLP kernel.
// One workgroup (256 threads = 8 wave32) owns 16 batch rows; cur1 and mem1
// tiles stay resident in LDS for all 25 timesteps. Layer-2 W2 fragments are
// held in registers across the whole time loop (loop-invariant).

typedef __attribute__((ext_vector_type(16))) __bf16 v16bf;
typedef __attribute__((ext_vector_type(8)))  __bf16 v8bf;
typedef __attribute__((ext_vector_type(8)))  float  v8f;

#define B_SZ   4096
#define IN_SZ  784
#define H_SZ   2048
#define OUT_SZ 10
#define ROWS   16      // batch rows per workgroup
#define PITCH1 2052    // f32 LDS pitch for [ROWS][H] tiles (+4 pad -> 16 distinct banks)
#define XPITCH 792     // bf16 LDS pitch for x tile (16B-aligned rows, bank-spread)
#define CPITCH 17      // f32 LDS pitch for 16x16 cur2 tile

struct alignas(16) SnnLds {
    float  cur1[ROWS * PITCH1];   // 131,328 B
    float  mem1[ROWS * PITCH1];   // 131,328 B
    __bf16 xs[ROWS * XPITCH];     //  25,344 B  (x tile pre-converted to bf16)
    float  cur2[ROWS * CPITCH];   //   1,088 B   -> total 289,088 B <= 320K
};

__global__ __launch_bounds__(256)
void snn_fused(const float* __restrict__ x, const float* __restrict__ W1,
               const float* __restrict__ W2, const float* __restrict__ beta1p,
               const float* __restrict__ beta2p, const int* __restrict__ nstepsp,
               float* __restrict__ out) {
    __shared__ SnnLds s;

    const int tid   = threadIdx.x;
    const int lane  = tid & 31;
    const int wave  = tid >> 5;
    const int m     = lane & 15;   // A row / B,C column index (ISA 16x16 layouts)
    const int khalf = lane >> 4;   // which K half-block this lane holds
    const int r0    = blockIdx.x * ROWS;

    const float beta1  = beta1p[0];
    const float beta2  = beta2p[0];
    const int   nsteps = nstepsp[0];

    // ---- startup: stage x tile (f32 -> bf16) in LDS; zero mem1 and cur2 ----
    for (int i = tid; i < ROWS * IN_SZ; i += 256) {
        const int row = i / IN_SZ, k = i - row * IN_SZ;
        s.xs[row * XPITCH + k] = (__bf16)x[(size_t)(r0 + row) * IN_SZ + k];
    }
    for (int i = tid; i < ROWS * PITCH1; i += 256) s.mem1[i] = 0.0f;
    for (int i = tid; i < ROWS * CPITCH; i += 256) s.cur2[i] = 0.0f;
    __syncthreads();

    // ---- phase 1: cur1 = x @ W1^T via v_wmma_f32_16x16x32_bf16 ----
    // Each wave produces 16 consecutive 16x16 output tiles (256 N-columns).
    {
        const __bf16* xr = s.xs + m * XPITCH;
        for (int tile = 0; tile < 16; ++tile) {
            const int n0 = wave * 256 + tile * 16;
            const float* w1r = W1 + (size_t)(n0 + m) * IN_SZ;
            v8f c = {};
            for (int kc = 0; kc < 25; ++kc) {        // K = 784 = 24*32 + 16
                const int kbase = kc * 32;
                v16bf a, b;
#pragma unroll
                for (int g = 0; g < 2; ++g) {
                    const int ks = kbase + g * 16 + khalf * 8;
                    if (ks < IN_SZ) {                // whole 8-group valid (784 % 16 == 0)
                        const v8bf av = *(const v8bf*)(xr + ks);   // one ds_load_b128
#pragma unroll
                        for (int j = 0; j < 8; ++j) a[g*8+j] = av[j];
                        const float4 h0 = *(const float4*)(w1r + ks);
                        const float4 h1 = *(const float4*)(w1r + ks + 4);
                        b[g*8+0]=(__bf16)h0.x; b[g*8+1]=(__bf16)h0.y;
                        b[g*8+2]=(__bf16)h0.z; b[g*8+3]=(__bf16)h0.w;
                        b[g*8+4]=(__bf16)h1.x; b[g*8+5]=(__bf16)h1.y;
                        b[g*8+6]=(__bf16)h1.z; b[g*8+7]=(__bf16)h1.w;
                    } else {
#pragma unroll
                        for (int j = 0; j < 8; ++j) { a[g*8+j] = (__bf16)0.0f; b[g*8+j] = (__bf16)0.0f; }
                    }
                }
                c = __builtin_amdgcn_wmma_f32_16x16x32_bf16(
                        false, a, false, b, (short)0, c, false, false);
            }
            // C layout: VGPR j -> M = j + 8*khalf, N = n0 + (lane&15)
#pragma unroll
            for (int j = 0; j < 8; ++j)
                s.cur1[(khalf * 8 + j) * PITCH1 + n0 + m] = c[j];
        }
    }

    // ---- load loop-invariant layer-2 B fragments (W2^T) into registers ----
    // B[k][n] = W2[n][k]; lane holds column n = lane&15 (cols 10..15 padded 0).
    const int  n      = lane & 15;
    const bool nvalid = (n < OUT_SZ);
    v16bf bw[8];
    {
        const float* w2r = W2 + (size_t)(nvalid ? n : 0) * H_SZ;
#pragma unroll
        for (int kc = 0; kc < 8; ++kc) {
#pragma unroll
            for (int g = 0; g < 2; ++g) {
                const int ks = wave * 256 + kc * 32 + g * 16 + khalf * 8;
                const float4 h0 = *(const float4*)(w2r + ks);
                const float4 h1 = *(const float4*)(w2r + ks + 4);
                bw[kc][g*8+0] = nvalid ? (__bf16)h0.x : (__bf16)0.0f;
                bw[kc][g*8+1] = nvalid ? (__bf16)h0.y : (__bf16)0.0f;
                bw[kc][g*8+2] = nvalid ? (__bf16)h0.z : (__bf16)0.0f;
                bw[kc][g*8+3] = nvalid ? (__bf16)h0.w : (__bf16)0.0f;
                bw[kc][g*8+4] = nvalid ? (__bf16)h1.x : (__bf16)0.0f;
                bw[kc][g*8+5] = nvalid ? (__bf16)h1.y : (__bf16)0.0f;
                bw[kc][g*8+6] = nvalid ? (__bf16)h1.z : (__bf16)0.0f;
                bw[kc][g*8+7] = nvalid ? (__bf16)h1.w : (__bf16)0.0f;
            }
        }
    }

    // ---- per-(row,out) layer-2 state in registers of threads 0..159 ----
    float mem2 = 0.0f, acc = 0.0f;
    const int crow = tid / OUT_SZ;
    const int cout = tid - crow * OUT_SZ;

    __syncthreads();

    // ---- time loop: 25 fused SNN steps entirely from LDS/registers ----
    for (int t = 0; t < nsteps; ++t) {
        // Phase A: leaky update of mem1 (vectorized float4); re-zero cur2
        for (int v = tid; v < ROWS * (H_SZ / 4); v += 256) {
            const int row  = v >> 9;           // H/4 = 512 float4 per row
            const int c4   = v & 511;
            const int addr = row * PITCH1 + c4 * 4;
            float4 mm = *(float4*)(s.mem1 + addr);
            const float4 cc = *(const float4*)(s.cur1 + addr);
            mm.x = beta1 * mm.x + cc.x - (mm.x > 1.0f ? 1.0f : 0.0f);
            mm.y = beta1 * mm.y + cc.y - (mm.y > 1.0f ? 1.0f : 0.0f);
            mm.z = beta1 * mm.z + cc.z - (mm.z > 1.0f ? 1.0f : 0.0f);
            mm.w = beta1 * mm.w + cc.w - (mm.w > 1.0f ? 1.0f : 0.0f);
            *(float4*)(s.mem1 + addr) = mm;
        }
        if (t > 0)  // cur2 already zeroed at startup for t==0
            for (int i = tid; i < ROWS * CPITCH; i += 256) s.cur2[i] = 0.0f;
        __syncthreads();

        // Phase B: cur2 += spike(mem1) @ W2^T ; 8 waves split K=2048 (256 each)
        {
            const float* mr = s.mem1 + m * PITCH1;
            v8f c2 = {};
#pragma unroll
            for (int kc = 0; kc < 8; ++kc) {
                const int kb = wave * 256 + kc * 32;
                v16bf a;
#pragma unroll
                for (int g = 0; g < 2; ++g) {
                    const int ks = kb + g * 16 + khalf * 8;
                    const float4 f0 = *(const float4*)(mr + ks);      // ds_load_b128
                    const float4 f1 = *(const float4*)(mr + ks + 4);  // ds_load_b128
                    a[g*8+0] = (f0.x > 1.0f) ? (__bf16)1.0f : (__bf16)0.0f;
                    a[g*8+1] = (f0.y > 1.0f) ? (__bf16)1.0f : (__bf16)0.0f;
                    a[g*8+2] = (f0.z > 1.0f) ? (__bf16)1.0f : (__bf16)0.0f;
                    a[g*8+3] = (f0.w > 1.0f) ? (__bf16)1.0f : (__bf16)0.0f;
                    a[g*8+4] = (f1.x > 1.0f) ? (__bf16)1.0f : (__bf16)0.0f;
                    a[g*8+5] = (f1.y > 1.0f) ? (__bf16)1.0f : (__bf16)0.0f;
                    a[g*8+6] = (f1.z > 1.0f) ? (__bf16)1.0f : (__bf16)0.0f;
                    a[g*8+7] = (f1.w > 1.0f) ? (__bf16)1.0f : (__bf16)0.0f;
                }
                c2 = __builtin_amdgcn_wmma_f32_16x16x32_bf16(
                        false, a, false, bw[kc], (short)0, c2, false, false);
            }
#pragma unroll
            for (int j = 0; j < 8; ++j)
                atomicAdd(&s.cur2[(khalf * 8 + j) * CPITCH + n], c2[j]);  // ds_add_f32
        }
        __syncthreads();

        // Phase C: layer-2 leaky update + spike-rate accumulation
        if (tid < ROWS * OUT_SZ) {
            const float cur2v = s.cur2[crow * CPITCH + cout];
            mem2 = beta2 * mem2 + cur2v - (mem2 > 1.0f ? 1.0f : 0.0f);
            acc += (mem2 > 1.0f) ? 1.0f : 0.0f;
        }
        __syncthreads();
    }

    if (tid < ROWS * OUT_SZ)
        out[(size_t)(r0 + crow) * OUT_SZ + cout] = acc * (1.0f / (float)nsteps);
}

extern "C" void kernel_launch(void* const* d_in, const int* in_sizes, int n_in,
                              void* d_out, int out_size, void* d_ws, size_t ws_size,
                              hipStream_t stream) {
    const float* x  = (const float*)d_in[0];
    const float* W1 = (const float*)d_in[1];
    const float* W2 = (const float*)d_in[2];
    const float* b1 = (const float*)d_in[3];
    const float* b2 = (const float*)d_in[4];
    const int*   ns = (const int*)d_in[5];
    float* out = (float*)d_out;

    dim3 grid(B_SZ / ROWS);   // 256 workgroups, one per 16-row batch tile
    dim3 block(256);          // 8 wave32
    hipLaunchKernelGGL(snn_fused, grid, block, 0, stream, x, W1, W2, b1, b2, ns, out);
}
